// FlashMultiHeadAttention_20555713478660
// MI455X (gfx1250) — matrix operand order repaired and checked by reading, not compiled, not run on hardware
//
#include <hip/hip_runtime.h>
#include <hip/hip_bf16.h>

typedef _Float16 f16;
typedef _Float16 v16h __attribute__((ext_vector_type(16)));
typedef _Float16 v8h  __attribute__((ext_vector_type(8)));
typedef float    v8f  __attribute__((ext_vector_type(8)));

constexpr int B_ = 4, T_ = 2048, D_ = 512, H_ = 8, DK_ = 64;
constexpr int BT = B_ * T_;
constexpr int D3 = 3 * D_;
constexpr float SCALE_  = 0.125f;   // 1/sqrt(64)
constexpr float LN_EPS_ = 1e-6f;
constexpr float NRM_EPS = 1e-8f;

// ---------------- WMMA helpers (CDNA5 16x16x32 f16 -> f32) ----------------
__device__ __forceinline__ v8f wmma16(v16h a, v16h b, v8f c) {
  // 8 args: (neg_a, A, neg_b, B, c_mod, C, reuse_a, reuse_b)
  return __builtin_amdgcn_wmma_f32_16x16x32_f16(false, a, false, b, (short)0, c,
                                                false, false);
}

// A-matrix fragment (16x32, M x K), lane: M = lane&15, hi = lane>>4.
// elems 0..7 -> K = e + 8*hi ; elems 8..15 -> K = 16 + (e-8) + 8*hi.
__device__ __forceinline__ v16h load_a16(const f16* p, int hi) {
  v8h lo = *(const v8h*)(p + hi * 8);
  v8h hh = *(const v8h*)(p + 16 + hi * 8);
  v16h r;
#pragma unroll
  for (int i = 0; i < 8; ++i) { r[i] = lo[i]; r[i + 8] = hh[i]; }
  return r;
}

// B-matrix fragment (32x16, K x N), lane: N = lane&15, hi = lane>>4.
// elems 0..7 -> K = e + 8*hi ; elems 8..15 -> K = 16 + (e-8) + 8*hi.
__device__ __forceinline__ v16h load_b16(const f16* p, int hi) {
  v8h lo = *(const v8h*)(p + hi * 8);
  v8h hh = *(const v8h*)(p + 16 + hi * 8);
  v16h r;
#pragma unroll
  for (int i = 0; i < 8; ++i) { r[i] = lo[i]; r[i + 8] = hh[i]; }
  return r;
}

// ---------------- weight conversion f32 -> f16 ----------------
__global__ __launch_bounds__(256) void cvt_kernel(const float* __restrict__ qkvw,
                                                  const float* __restrict__ outw,
                                                  f16* __restrict__ wq,
                                                  f16* __restrict__ wo) {
  int i = blockIdx.x * 256 + threadIdx.x;
  if (i < D3 * D_) wq[i] = (f16)qkvw[i];
  if (i < D_ * D_) wo[i] = (f16)outw[i];
}

// ---------------- LayerNorm -> f16 ----------------
__global__ __launch_bounds__(256) void ln_kernel(const float* __restrict__ x,
                                                 const float* __restrict__ g,
                                                 const float* __restrict__ bb,
                                                 f16* __restrict__ xn) {
  int row  = blockIdx.x * 8 + (threadIdx.x >> 5);
  int lane = threadIdx.x & 31;
  const float* xr = x + (size_t)row * D_;
  float vals[16];
  float s = 0.f, ss = 0.f;
#pragma unroll
  for (int i = 0; i < 16; ++i) {
    float v = xr[lane + 32 * i];
    vals[i] = v; s += v; ss += v * v;
  }
#pragma unroll
  for (int off = 1; off < 32; off <<= 1) {
    s  += __shfl_xor(s, off, 32);
    ss += __shfl_xor(ss, off, 32);
  }
  float mu  = s * (1.0f / D_);
  float var = ss * (1.0f / D_) - mu * mu;
  float rs  = rsqrtf(var + LN_EPS_);
  f16* xo = xn + (size_t)row * D_;
#pragma unroll
  for (int i = 0; i < 16; ++i) {
    int d = lane + 32 * i;
    xo[d] = (f16)((vals[i] - mu) * rs * g[d] + bb[d]);
  }
}

// ------- generic GEMM: C[M,N] = A[M,K] * W[N,K]^T + bias, 32x64 per wave ---
template <bool F32OUT>
__global__ __launch_bounds__(256) void gemm_kernel(const f16* __restrict__ A,
                                                   const f16* __restrict__ W,
                                                   const float* __restrict__ bias,
                                                   void* __restrict__ out,
                                                   int M, int N, int K, float scale) {
  int wid  = (int)((blockIdx.x * blockDim.x + threadIdx.x) >> 5);
  int lane = threadIdx.x & 31;
  int ln15 = lane & 15, hi = lane >> 4;
  int ngroups = N >> 6;                  // wave owns a 32(M) x 64(N) block
  int total   = (M >> 5) * ngroups;
  if (wid >= total) return;
  int mt = wid / ngroups, ng = wid % ngroups;

  const f16* arow0 = A + (size_t)(mt * 32 + ln15) * K;
  const f16* arow1 = arow0 + (size_t)16 * K;
  const f16* wg    = W + (size_t)(ng * 64 + ln15) * K;
  v8f acc[2][4] = {};
  for (int k0 = 0; k0 < K; k0 += 32) {
    v16h a0 = load_a16(arow0 + k0, hi);
    v16h a1 = load_a16(arow1 + k0, hi);
#pragma unroll
    for (int nt = 0; nt < 4; ++nt) {
      v16h b = load_b16(wg + (size_t)nt * 16 * K + k0, hi);
      acc[0][nt] = wmma16(a0, b, acc[0][nt]);
      acc[1][nt] = wmma16(a1, b, acc[1][nt]);
    }
  }
#pragma unroll
  for (int mu = 0; mu < 2; ++mu) {
#pragma unroll
    for (int nt = 0; nt < 4; ++nt) {
      int col = ng * 64 + nt * 16 + ln15;
      float bv = bias[col];
#pragma unroll
      for (int r = 0; r < 8; ++r) {
        int m = mt * 32 + mu * 16 + r + 8 * hi;   // D-layout rows
        float v = (acc[mu][nt][r] + bv) * scale;
        if (F32OUT) ((float*)out)[(size_t)m * N + col] = v;
        else        ((f16*)out)[(size_t)m * N + col]   = (f16)v;
      }
    }
  }
}

// ---------------- L2-norm Q,K in place + transpose V ----------------
__global__ __launch_bounds__(256) void pack_kernel(f16* __restrict__ qkv,
                                                   f16* __restrict__ vt) {
  int rid  = blockIdx.x * 8 + (threadIdx.x >> 5);
  int lane = threadIdx.x & 31;
  int t  = rid % T_;
  int bh = rid / T_;
  int h = bh % H_, b = bh / H_;
  f16* base = qkv + (size_t)(b * T_ + t) * D3 + h * DK_;
  {   // Q
    float q0 = (float)base[lane], q1 = (float)base[lane + 32];
    float ss = q0 * q0 + q1 * q1;
#pragma unroll
    for (int off = 1; off < 32; off <<= 1) ss += __shfl_xor(ss, off, 32);
    float sc = 1.0f / fmaxf(sqrtf(ss), NRM_EPS);
    base[lane] = (f16)(q0 * sc); base[lane + 32] = (f16)(q1 * sc);
  }
  {   // K
    f16* kb = base + D_;
    float k0 = (float)kb[lane], k1 = (float)kb[lane + 32];
    float ss = k0 * k0 + k1 * k1;
#pragma unroll
    for (int off = 1; off < 32; off <<= 1) ss += __shfl_xor(ss, off, 32);
    float sc = 1.0f / fmaxf(sqrtf(ss), NRM_EPS);
    kb[lane] = (f16)(k0 * sc); kb[lane + 32] = (f16)(k1 * sc);
  }
  {   // V -> Vt[b,h,dk,t]
    const f16* vb = base + 2 * D_;
    f16* vrow = vt + (size_t)(b * H_ + h) * DK_ * T_ + t;
    vrow[(size_t)lane * T_]        = vb[lane];
    vrow[(size_t)(lane + 32) * T_] = vb[lane + 32];
  }
}

// ------- flash attention: 1 wave per (b,h, 32-query block), 2 q-tiles -----
__global__ __launch_bounds__(32) void attn_kernel(const f16* __restrict__ qkv,
                                                  const f16* __restrict__ vt,
                                                  const int* __restrict__ mask,
                                                  f16* __restrict__ ao) {
  __shared__ __align__(16) f16 pbuf[32 * 32];   // P tiles bounce (D->A layout)
  const int lane = threadIdx.x & 31;
  const int ln15 = lane & 15;
  const int hi   = lane >> 4;
  const int TQ = T_ / 32;
  int bh = blockIdx.x / TQ;
  int qt = blockIdx.x % TQ;
  int b = bh / H_, h = bh % H_;
  int qb = qt * 32;

  v16h a0[2], a1[2];
#pragma unroll
  for (int u = 0; u < 2; ++u) {
    const f16* qrow = qkv + (size_t)(b * T_ + qb + u * 16 + ln15) * D3 + h * DK_;
    a0[u] = load_a16(qrow, hi);        // Q d=0..31
    a1[u] = load_a16(qrow + 32, hi);   // Q d=32..63
  }

  v8f o[2][4] = {};
  float rmax[2][8], rsum[2][8];
#pragma unroll
  for (int u = 0; u < 2; ++u)
#pragma unroll
    for (int r = 0; r < 8; ++r) { rmax[u][r] = -1e30f; rsum[u][r] = 0.f; }

  const f16* kbase = qkv + (size_t)b * T_ * D3 + D_ + h * DK_;
  const f16* vbase = vt + (size_t)(b * H_ + h) * DK_ * T_;
  const int* mrow  = mask + b * T_;

  for (int j = 0; j < T_; j += 32) {
    int key0 = j + ln15, key1 = key0 + 16;
    const f16* k0p = kbase + (size_t)key0 * D3;
    const f16* k1p = kbase + (size_t)key1 * D3;
    v16h kb00 = load_b16(k0p, hi);
    v16h kb01 = load_b16(k0p + 32, hi);
    v16h kb10 = load_b16(k1p, hi);
    v16h kb11 = load_b16(k1p + 32, hi);
    bool m0 = mrow[key0] != 0;
    bool m1 = mrow[key1] != 0;

#pragma unroll
    for (int u = 0; u < 2; ++u) {
      v8f zc = {};
      v8f s0 = wmma16(a0[u], kb00, zc);  s0 = wmma16(a1[u], kb01, s0);
      v8f s1 = wmma16(a0[u], kb10, zc);  s1 = wmma16(a1[u], kb11, s1);
#pragma unroll
      for (int r = 0; r < 8; ++r) {
        float x0 = fminf(fmaxf(s0[r] * SCALE_, -10.f), 10.f);
        float x1 = fminf(fmaxf(s1[r] * SCALE_, -10.f), 10.f);
        x0 = m0 ? -10000.f : x0;
        x1 = m1 ? -10000.f : x1;
        float tm = fmaxf(x0, x1);
#pragma unroll
        for (int off = 1; off < 16; off <<= 1)      // stays inside lane-half
          tm = fmaxf(tm, __shfl_xor(tm, off, 32));
        float nm   = fmaxf(rmax[u][r], tm);
        float corr = __expf(rmax[u][r] - nm);
        float p0 = __expf(x0 - nm);
        float p1 = __expf(x1 - nm);
        float ts = p0 + p1;
#pragma unroll
        for (int off = 1; off < 16; off <<= 1)
          ts += __shfl_xor(ts, off, 32);
        rsum[u][r] = rsum[u][r] * corr + ts;
        rmax[u][r] = nm;
        o[u][0][r] *= corr; o[u][1][r] *= corr;
        o[u][2][r] *= corr; o[u][3][r] *= corr;
        int m = u * 16 + r + 8 * hi;     // D-layout row this lane-half holds
        pbuf[m * 32 + ln15]      = (f16)p0;
        pbuf[m * 32 + 16 + ln15] = (f16)p1;
      }
    }
    __syncthreads();
    v16h pa0 = load_a16(&pbuf[ln15 * 32], hi);          // P tile u=0
    v16h pa1 = load_a16(&pbuf[(16 + ln15) * 32], hi);   // P tile u=1
    const f16* vj = vbase + j;
#pragma unroll
    for (int nt = 0; nt < 4; ++nt) {
      v16h vb = load_b16(vj + (size_t)(nt * 16 + ln15) * T_, hi);
      o[0][nt] = wmma16(pa0, vb, o[0][nt]);
      o[1][nt] = wmma16(pa1, vb, o[1][nt]);
    }
    __syncthreads();
  }
#pragma unroll
  for (int u = 0; u < 2; ++u) {
#pragma unroll
    for (int r = 0; r < 8; ++r) {
      float inv = 1.0f / rsum[u][r];
      int m = u * 16 + r + 8 * hi;
      size_t row = (size_t)(b * T_ + qb + m) * D_ + h * DK_;
#pragma unroll
      for (int nt = 0; nt < 4; ++nt)
        ao[row + nt * 16 + ln15] = (f16)(o[u][nt][r] * inv);
    }
  }
}

// ---------------- launch ----------------
extern "C" void kernel_launch(void* const* d_in, const int* in_sizes, int n_in,
                              void* d_out, int out_size, void* d_ws, size_t ws_size,
                              hipStream_t stream) {
  const float* x     = (const float*)d_in[0];
  const int*   mask  = (const int*)d_in[1];
  const float* ln_g  = (const float*)d_in[2];
  const float* ln_b  = (const float*)d_in[3];
  const float* qkv_w = (const float*)d_in[4];
  const float* qkv_b = (const float*)d_in[5];
  const float* out_w = (const float*)d_in[6];
  const float* out_b = (const float*)d_in[7];
  float* out = (float*)d_out;

  char* ws = (char*)d_ws;
  // layout (bytes): XN/AO 8 MB | WQKV 1.5 MB | WOUT 0.5 MB | QKV 24 MB | VT 8 MB
  f16* XN   = (f16*)ws;                                  // BT*D f16 (reused as AO)
  f16* WQKV = (f16*)(ws + (size_t)BT * D_ * 2);
  f16* WOUT = (f16*)(ws + (size_t)BT * D_ * 2 + (size_t)D3 * D_ * 2);
  f16* QKV  = (f16*)(ws + (size_t)BT * D_ * 2 + (size_t)D3 * D_ * 2 + (size_t)D_ * D_ * 2);
  f16* VT   = (f16*)((char*)QKV + (size_t)BT * D3 * 2);
  f16* AO   = XN;   // XN dead after qkv GEMM

  cvt_kernel<<<(D3 * D_ + 255) / 256, 256, 0, stream>>>(qkv_w, out_w, WQKV, WOUT);
  ln_kernel<<<BT / 8, 256, 0, stream>>>(x, ln_g, ln_b, XN);
  gemm_kernel<false><<<(BT / 32) * (D3 / 64) / 8, 256, 0, stream>>>(
      XN, WQKV, qkv_b, QKV, BT, D3, D_, 1.0f);
  pack_kernel<<<(B_ * H_ * T_) / 8, 256, 0, stream>>>(QKV, VT);
  attn_kernel<<<B_ * H_ * (T_ / 32), 32, 0, stream>>>(QKV, VT, mask, AO);
  gemm_kernel<true><<<(BT / 32) * (D_ / 64) / 8, 256, 0, stream>>>(
      AO, WOUT, out_b, out, BT, D_, D_, 0.5f);
}
